// GraphSAGE_6330781794594
// MI455X (gfx1250) — matrix operand order repaired and checked by reading, not compiled, not run on hardware
//
#include <hip/hip_runtime.h>
#include <hip/hip_fp16.h>
#include <math.h>

#define N_NODES 100000
#define N_EDGES 3200000
#define D_IN    128
#define D_HID   128
#define D_OUT   40

typedef __attribute__((ext_vector_type(16))) _Float16 v16h;
typedef __attribute__((ext_vector_type(8)))  float    v8f;

// ---------------------------------------------------------------------------
// A-fragment loader: 16x32 f16 A tile from row-major f32 matrix [*, lda].
// Per ISA table (16-bit A 16x32, wave32):
//   lanes 0-15 : row M = lane,    elems 0..7 -> K = k0+0..7,  8..15 -> K = k0+16..23
//   lanes 16-31: row M = lane-16, elems 0..7 -> K = k0+8..15, 8..15 -> K = k0+24..31
// Scale applied on load (mean-aggregation normalization fused here).
// ---------------------------------------------------------------------------
__device__ __forceinline__ v16h load_a_f32(const float* __restrict__ A, int lda,
                                           int m0, int k0, int lane, float scale) {
    int m = m0 + (lane & 15);
    int k = k0 + ((lane & 16) ? 8 : 0);
    const float* p = A + (size_t)m * lda + k;
    v16h a;
#pragma unroll
    for (int e = 0; e < 8; ++e) a[e] = (_Float16)(p[e] * scale);
#pragma unroll
    for (int e = 0; e < 8; ++e) a[8 + e] = (_Float16)(p[16 + e] * scale);
    return a;
}

// ---------------------------------------------------------------------------
// Weight pre-pack: convert row-major f32 W[K=128, N] into f16 fragments laid
// out so each lane's 16 B-elements are one contiguous, 32B-aligned chunk:
//   Wpk[((kt*ntiles + nt)*32 + lane)*16 + e]
// B layout (32x16, wave32): lanes 0-15: col N = n0+lane,    e 0..7 -> K = k0+e,   8..15 -> K = k0+16+(e-8)
//                           lanes 16-31: col N = n0+lane-16, e 0..7 -> K = k0+8+e, 8..15 -> K = k0+24+(e-8)
// Columns >= nmax zero-filled (N=40 output layer padded to 48).
// One thread packs one lane-fragment (16 f16).
// ---------------------------------------------------------------------------
__global__ __launch_bounds__(256) void pack_b_kernel(const float* __restrict__ W,
                                                     int nmax, int ktiles, int ntiles,
                                                     _Float16* __restrict__ out) {
    int t = blockIdx.x * blockDim.x + threadIdx.x;
    if (t >= ktiles * ntiles * 32) return;
    int lane = t & 31;
    int frag = t >> 5;
    int kt = frag / ntiles;
    int nt = frag % ntiles;
    int k = kt * 32 + ((lane & 16) ? 8 : 0);
    int n = nt * 16 + (lane & 15);
    _Float16* o = out + (size_t)t * 16;
#pragma unroll
    for (int e = 0; e < 16; ++e)
        o[e] = (n < nmax) ? (_Float16)W[(size_t)(k + e + (e >= 8 ? 8 : 0)) * nmax + n] : (_Float16)0.0f;
}

// ---------------------------------------------------------------------------
// Degree: one thread per edge, f32 atomic inc at dst; then clamped reciprocal.
// ---------------------------------------------------------------------------
__global__ __launch_bounds__(256) void deg_kernel(const int* __restrict__ dst,
                                                  float* __restrict__ deg) {
    int e = blockIdx.x * blockDim.x + threadIdx.x;
    if (e < N_EDGES) unsafeAtomicAdd(&deg[dst[e]], 1.0f);
}

__global__ __launch_bounds__(256) void inv_kernel(float* __restrict__ deg) {
    int i = blockIdx.x * blockDim.x + threadIdx.x;
    if (i < N_NODES) deg[i] = 1.0f / fmaxf(deg[i], 1.0f);
}

// ---------------------------------------------------------------------------
// Scatter-add: one wave per edge. 32 lanes x float4 = full 128-float row,
// coalesced 512B read of feat[src], f32 atomics into agg[dst] (L2-resident).
// ---------------------------------------------------------------------------
__global__ __launch_bounds__(256) void scatter_kernel(const float* __restrict__ feat,
                                                      const int* __restrict__ src,
                                                      const int* __restrict__ dst,
                                                      float* __restrict__ agg) {
    int wave = (blockIdx.x * blockDim.x + threadIdx.x) >> 5;
    int lane = threadIdx.x & 31;
    if (wave >= N_EDGES) return;
    int s = src[wave];
    int d = dst[wave];
    float4 v = *((const float4*)(feat + (size_t)s * D_IN) + lane);
    float* ap = agg + (size_t)d * D_IN + lane * 4;
    unsafeAtomicAdd(ap + 0, v.x);
    unsafeAtomicAdd(ap + 1, v.y);
    unsafeAtomicAdd(ap + 2, v.z);
    unsafeAtomicAdd(ap + 3, v.w);
}

// ---------------------------------------------------------------------------
// Layer 1 fused GEMM: h = relu((agg .* inv) @ W_l + x @ W_r + bias)
// One wave owns a full 16x128 output row-block: each A fragment is loaded
// once and multiplied against 8 pre-packed B fragments (8 accumulators).
// 64 v_wmma_f32_16x16x32_f16 per wave; K = 256 effective.
// ---------------------------------------------------------------------------
__global__ __launch_bounds__(256) void gemm1_kernel(const float* __restrict__ agg,
                                                    const float* __restrict__ x,
                                                    const v16h* __restrict__ Bl,
                                                    const v16h* __restrict__ Br,
                                                    const float* __restrict__ bias,
                                                    const float* __restrict__ inv,
                                                    float* __restrict__ h) {
    int wave = (blockIdx.x * blockDim.x + threadIdx.x) >> 5;
    int lane = threadIdx.x & 31;
    if (wave >= N_NODES / 16) return;
    int m0 = wave * 16;
    float scale = inv[m0 + (lane & 15)];

    v8f acc[8];
#pragma unroll
    for (int nt = 0; nt < 8; ++nt) acc[nt] = (v8f){};

#pragma unroll
    for (int kc = 0; kc < 4; ++kc) {            // mean(agg) @ W_l, K = 128
        v16h a = load_a_f32(agg, D_IN, m0, kc * 32, lane, scale);
#pragma unroll
        for (int nt = 0; nt < 8; ++nt) {
            v16h b = Bl[(kc * 8 + nt) * 32 + lane];
            acc[nt] = __builtin_amdgcn_wmma_f32_16x16x32_f16(false, a, false, b,
                                                             (short)0, acc[nt],
                                                             false, false);
        }
    }
#pragma unroll
    for (int kc = 0; kc < 4; ++kc) {            // x @ W_r, K = 128
        v16h a = load_a_f32(x, D_IN, m0, kc * 32, lane, 1.0f);
#pragma unroll
        for (int nt = 0; nt < 8; ++nt) {
            v16h b = Br[(kc * 8 + nt) * 32 + lane];
            acc[nt] = __builtin_amdgcn_wmma_f32_16x16x32_f16(false, a, false, b,
                                                             (short)0, acc[nt],
                                                             false, false);
        }
    }

    int mrow = m0 + ((lane & 16) ? 8 : 0);
#pragma unroll
    for (int nt = 0; nt < 8; ++nt) {
        int n = nt * 16 + (lane & 15);
        float bv = bias[n];
#pragma unroll
        for (int r = 0; r < 8; ++r) {
            float v = acc[nt][r] + bv;
            h[(size_t)(mrow + r) * D_HID + n] = v > 0.0f ? v : 0.0f;
        }
    }
}

// ---------------------------------------------------------------------------
// Layer 2 fused GEMM: out = (agg2 .* inv) @ W2_l + h @ W2_r + b2  (N = 40)
// 3 N-tiles (cols 40..47 zero-padded in packed B, stores masked).
// ---------------------------------------------------------------------------
__global__ __launch_bounds__(256) void gemm2_kernel(const float* __restrict__ agg,
                                                    const float* __restrict__ hfeat,
                                                    const v16h* __restrict__ Bl,
                                                    const v16h* __restrict__ Br,
                                                    const float* __restrict__ bias,
                                                    const float* __restrict__ inv,
                                                    float* __restrict__ out) {
    int wave = (blockIdx.x * blockDim.x + threadIdx.x) >> 5;
    int lane = threadIdx.x & 31;
    if (wave >= N_NODES / 16) return;
    int m0 = wave * 16;
    float scale = inv[m0 + (lane & 15)];

    v8f acc[3];
#pragma unroll
    for (int nt = 0; nt < 3; ++nt) acc[nt] = (v8f){};

#pragma unroll
    for (int kc = 0; kc < 4; ++kc) {
        v16h a = load_a_f32(agg, D_HID, m0, kc * 32, lane, scale);
#pragma unroll
        for (int nt = 0; nt < 3; ++nt) {
            v16h b = Bl[(kc * 3 + nt) * 32 + lane];
            acc[nt] = __builtin_amdgcn_wmma_f32_16x16x32_f16(false, a, false, b,
                                                             (short)0, acc[nt],
                                                             false, false);
        }
    }
#pragma unroll
    for (int kc = 0; kc < 4; ++kc) {
        v16h a = load_a_f32(hfeat, D_HID, m0, kc * 32, lane, 1.0f);
#pragma unroll
        for (int nt = 0; nt < 3; ++nt) {
            v16h b = Br[(kc * 3 + nt) * 32 + lane];
            acc[nt] = __builtin_amdgcn_wmma_f32_16x16x32_f16(false, a, false, b,
                                                             (short)0, acc[nt],
                                                             false, false);
        }
    }

    int mrow = m0 + ((lane & 16) ? 8 : 0);
#pragma unroll
    for (int nt = 0; nt < 3; ++nt) {
        int n = nt * 16 + (lane & 15);
        if (n < D_OUT) {
            float bv = bias[n];
#pragma unroll
            for (int r = 0; r < 8; ++r)
                out[(size_t)(mrow + r) * D_OUT + n] = acc[nt][r] + bv;
        }
    }
}

// ---------------------------------------------------------------------------
// log_softmax over 40 cols, one wave per row, in place on d_out.
// ---------------------------------------------------------------------------
__global__ __launch_bounds__(256) void logsoftmax_kernel(float* __restrict__ out) {
    int wave = (blockIdx.x * blockDim.x + threadIdx.x) >> 5;
    int lane = threadIdx.x & 31;
    if (wave >= N_NODES) return;
    float* row = out + (size_t)wave * D_OUT;
    float v0 = (lane < D_OUT) ? row[lane] : -__builtin_inff();
    float v1 = (lane + 32 < D_OUT) ? row[lane + 32] : -__builtin_inff();
    float m = fmaxf(v0, v1);
#pragma unroll
    for (int off = 16; off > 0; off >>= 1) m = fmaxf(m, __shfl_xor(m, off, 32));
    float s = ((lane < D_OUT) ? __expf(v0 - m) : 0.0f) +
              ((lane + 32 < D_OUT) ? __expf(v1 - m) : 0.0f);
#pragma unroll
    for (int off = 16; off > 0; off >>= 1) s += __shfl_xor(s, off, 32);
    float lse = m + __logf(s);
    if (lane < D_OUT) row[lane] = v0 - lse;
    if (lane + 32 < D_OUT) row[lane + 32] = v1 - lse;
}

// ---------------------------------------------------------------------------
extern "C" void kernel_launch(void* const* d_in, const int* in_sizes, int n_in,
                              void* d_out, int out_size, void* d_ws, size_t ws_size,
                              hipStream_t stream) {
    const float* x   = (const float*)d_in[0];
    const int*   ei  = (const int*)d_in[1];          // edge_index [2, E]
    const float* W1l = (const float*)d_in[2];
    const float* W1r = (const float*)d_in[3];
    const float* b1  = (const float*)d_in[4];
    const float* W2l = (const float*)d_in[5];
    const float* W2r = (const float*)d_in[6];
    const float* b2  = (const float*)d_in[7];
    const int* src = ei;
    const int* dst = ei + N_EDGES;
    float* out = (float*)d_out;

    const size_t FEAT_BYTES = (size_t)N_NODES * D_IN * sizeof(float);  // 51.2 MB
    const size_t PK1_ELEMS  = 4 * 8 * 32 * 16;   // 128x128 packed: 16384 f16
    const size_t PK2_ELEMS  = 4 * 3 * 32 * 16;   // 128x40(->48) packed: 6144 f16

    char* ws = (char*)d_ws;
    float*    inv   = (float*)ws;                                    // [N]
    float*    agg   = (float*)(ws + (1 << 20));                      // [N,128]
    float*    h     = (float*)(ws + (1 << 20) + FEAT_BYTES);         // [N,128]
    _Float16* pk1l  = (_Float16*)(ws + (1 << 20) + 2 * FEAT_BYTES);
    _Float16* pk1r  = pk1l + PK1_ELEMS;
    _Float16* pk2l  = pk1r + PK1_ELEMS;
    _Float16* pk2r  = pk2l + PK2_ELEMS;

    // one-time weight packing (tiny)
    pack_b_kernel<<<(4 * 8 * 32 + 255) / 256, 256, 0, stream>>>(W1l, D_HID, 4, 8, pk1l);
    pack_b_kernel<<<(4 * 8 * 32 + 255) / 256, 256, 0, stream>>>(W1r, D_HID, 4, 8, pk1r);
    pack_b_kernel<<<(4 * 3 * 32 + 255) / 256, 256, 0, stream>>>(W2l, D_OUT, 4, 3, pk2l);
    pack_b_kernel<<<(4 * 3 * 32 + 255) / 256, 256, 0, stream>>>(W2r, D_OUT, 4, 3, pk2r);

    // degree + clamped reciprocal
    hipMemsetAsync(inv, 0, (size_t)N_NODES * sizeof(float), stream);
    hipMemsetAsync(agg, 0, FEAT_BYTES, stream);
    deg_kernel<<<(N_EDGES + 255) / 256, 256, 0, stream>>>(dst, inv);
    inv_kernel<<<(N_NODES + 255) / 256, 256, 0, stream>>>(inv);

    // layer 1: scatter x, fused WMMA GEMM + bias + relu
    scatter_kernel<<<N_EDGES / 8, 256, 0, stream>>>(x, src, dst, agg);
    gemm1_kernel<<<(N_NODES / 16 + 7) / 8, 256, 0, stream>>>(agg, x, (const v16h*)pk1l,
                                                             (const v16h*)pk1r, b1, inv, h);

    // layer 2: re-zero accumulator, scatter h, fused WMMA GEMM + bias
    hipMemsetAsync(agg, 0, FEAT_BYTES, stream);
    scatter_kernel<<<N_EDGES / 8, 256, 0, stream>>>(h, src, dst, agg);
    gemm2_kernel<<<(N_NODES / 16 + 7) / 8, 256, 0, stream>>>(agg, h, (const v16h*)pk2l,
                                                             (const v16h*)pk2r, b2, inv, out);

    // log-softmax in place on the output
    logsoftmax_kernel<<<N_NODES / 8, 256, 0, stream>>>(out);
}
